// MHSA_16621523436271
// MI455X (gfx1250) — matrix operand order repaired and checked by reading, not compiled, not run on hardware
//
#include <hip/hip_runtime.h>
#include <hip/hip_bf16.h>

// Problem constants (match reference)
#define DIM   768
#define NH    12
#define DH    64
#define SEQ   2048
#define BATCH 2
#define MTOT  (BATCH * SEQ)   // 4096
#define QKVN  (3 * DIM)       // 2304

typedef __attribute__((ext_vector_type(16))) __bf16 v16bf;
typedef __attribute__((ext_vector_type(8)))  __bf16 v8bf;
typedef __attribute__((ext_vector_type(8)))  float  v8f;

static __device__ __forceinline__ v16bf cat16(v8bf lo, v8bf hi) {
  v16bf r;
#pragma unroll
  for (int i = 0; i < 8; ++i) { r[i] = lo[i]; r[i + 8] = hi[i]; }
  return r;
}

// CDNA5 async global->LDS DMA (ASYNCcnt-tracked), 16B per lane.
// VDST = LDS byte address (low 32 bits of the flat LDS pointer == LDS offset
// per the aperture mapping), VADDR = 64-bit global address, SADDR = off.
static __device__ __forceinline__ void async_ld16(unsigned lds, const void* g) {
  asm volatile("global_load_async_to_lds_b128 %0, %1, off"
               :: "v"(lds), "v"(g) : "memory");
}
static __device__ __forceinline__ void wait_async0() {
  asm volatile("s_wait_asynccnt 0x0" ::: "memory");
}

// ---------------- conversion kernels ----------------

__global__ void cvt_f32_bf16(const float* __restrict__ src,
                             __bf16* __restrict__ dst, int n) {
  for (int i = blockIdx.x * 256 + threadIdx.x; i < n; i += gridDim.x * 256)
    dst[i] = (__bf16)src[i];
}

// w[K][Nt] (row-major) -> wt[Nt][K] (transposed bf16) so B-fragments
// (lane = output column, contiguous contraction index) are unit-stride.
__global__ void cvt_w_transpose(const float* __restrict__ w,
                                __bf16* __restrict__ wt, int K, int Nt) {
  int total = K * Nt;
  for (int i = blockIdx.x * 256 + threadIdx.x; i < total; i += gridDim.x * 256) {
    int k = i / Nt, n = i - k * Nt;
    wt[n * K + k] = (__bf16)w[i];
  }
}

// ---------------- tiled WMMA GEMM (async double-buffered) ----------------
// C[M][Ntot] = A[M][768] * Bt[Ntot][768]^T + bias
// Block tile 128x128, 8 waves of 64x32 (8 wmma / 12 ds_load_b128 per K-step).
// Tiles staged with global_load_async_to_lds_b128, ping-pong LDS buffers,
// one barrier + s_wait_asynccnt per K-step.
// mode 0: fp32 out.  mode 1: scatter Q/K ([bh][n][dh]) and V transposed
// ([bh][dh][n]) in bf16.
#define TM 128
#define TN 128
#define TK 32
#define LPAD 40   // LDS row stride (bf16): 80B -> conflict-free fragment reads

__global__ __launch_bounds__(256)
void gemm_bf16(const __bf16* __restrict__ A, const __bf16* __restrict__ Bt,
               const float* __restrict__ bias, int Ntot, int mode,
               float* __restrict__ Cout,
               __bf16* __restrict__ Qd, __bf16* __restrict__ Kd,
               __bf16* __restrict__ Vt) {
  __shared__ __bf16 As[2][TM][LPAD];   // 2 x 10 KB
  __shared__ __bf16 Bs[2][TN][LPAD];   // 2 x 10 KB

  const int K = DIM;
  const int tiles_n = Ntot / TN;
  const int m0 = (blockIdx.x / tiles_n) * TM;
  const int n0 = (blockIdx.x % tiles_n) * TN;
  const int tid  = threadIdx.x;
  const int lane = tid & 31;
  const int w    = tid >> 5;
  const int wm = (w & 1) * 64;   // wave M offset (2 waves in M)
  const int wn = (w >> 1) * 32;  // wave N offset (4 waves in N)

  v8f acc[4][2] = {};

  const int lr    = lane & 15;
  const int koffA = (lane >> 4) * 16;  // A frag: lanes 16-31 hold K=16..31
  const int koffB = (lane >> 4) * 16;  // B frag: lanes 16-31 hold K=16..31

  // Tile staging: 128x32 bf16 = 512 16B chunks per matrix; 2 chunks/thread.
  // chunk c -> row = c>>2, 16B segment = c&3.
  auto stage = [&](int buf, int k0) {
#pragma unroll
    for (int i = 0; i < 2; ++i) {
      const int c = tid + i * 256;
      const int row = c >> 2, seg = c & 3;
      async_ld16((unsigned)(size_t)&As[buf][row][seg * 8],
                 (const void*)(A + (size_t)(m0 + row) * K + k0 + seg * 8));
      async_ld16((unsigned)(size_t)&Bs[buf][row][seg * 8],
                 (const void*)(Bt + (size_t)(n0 + row) * K + k0 + seg * 8));
    }
  };

  stage(0, 0);
  wait_async0();
  __syncthreads();

  const int nk = K / TK;   // 24
  for (int kt = 0; kt < nk; ++kt) {
    const int cur = kt & 1;
    if (kt + 1 < nk) stage(cur ^ 1, (kt + 1) * TK);

    const __bf16(*Ac)[LPAD] = As[cur];
    const __bf16(*Bc)[LPAD] = Bs[cur];

    v16bf bfrag[2];
#pragma unroll
    for (int ni = 0; ni < 2; ++ni) {
      const __bf16* bp = &Bc[wn + ni * 16 + lr][koffB];
      bfrag[ni] = cat16(*(const v8bf*)bp, *(const v8bf*)(bp + 8));
    }
#pragma unroll
    for (int mi = 0; mi < 4; ++mi) {
      const __bf16* ap = &Ac[wm + mi * 16 + lr][0];
      v16bf afrag = cat16(*(const v8bf*)(ap + koffA),
                          *(const v8bf*)(ap + koffA + 8));
#pragma unroll
      for (int ni = 0; ni < 2; ++ni)
        acc[mi][ni] = __builtin_amdgcn_wmma_f32_16x16x32_bf16(
            false, afrag, false, bfrag[ni], (short)0, acc[mi][ni], false, false);
    }

    wait_async0();     // our next-tile DMAs have landed
    __syncthreads();   // everyone done reading cur / writing next
  }

  // Epilogue. C-layout: lane holds column (lane&15); VGPR r = row r+(lane>=16)*8.
  const int cn  = lane & 15;
  const int mhi = (lane >> 4) * 8;
#pragma unroll
  for (int mi = 0; mi < 4; ++mi) {
#pragma unroll
    for (int ni = 0; ni < 2; ++ni) {
      const int gn = n0 + wn + ni * 16 + cn;
      const float bv = bias[gn];
#pragma unroll
      for (int r = 0; r < 8; ++r) {
        const int gm = m0 + wm + mi * 16 + mhi + r;
        const float v = acc[mi][ni][r] + bv;
        if (mode == 0) {
          Cout[(size_t)gm * Ntot + gn] = v;
        } else {
          const int which = gn / DIM;          // 0=Q 1=K 2=V
          const int d  = gn - which * DIM;
          const int h  = d >> 6, dh = d & 63;
          const int b  = gm >> 11, sn = gm & 2047;
          const int bh = b * NH + h;
          const __bf16 bf = (__bf16)v;
          if (which == 0)      Qd[((size_t)bh * SEQ + sn) * DH + dh] = bf;
          else if (which == 1) Kd[((size_t)bh * SEQ + sn) * DH + dh] = bf;
          else                 Vt[((size_t)bh * DH + dh) * SEQ + sn] = bf;
        }
      }
    }
  }
}

// ---------------- flash attention ----------------
// One wave per 16-row q tile; 8 waves (128 q rows) per block; keys in chunks
// of 32 with online softmax. K/V stream from global (fits 192MB L2), P
// transposes C-layout -> A-layout through a per-wave LDS patch.
#define QTILE  16
#define KCHUNK 32
#define WAVES  8

__global__ __launch_bounds__(256)
void flash_attn(const __bf16* __restrict__ Qd, const __bf16* __restrict__ Kd,
                const __bf16* __restrict__ Vt, __bf16* __restrict__ Out) {
  __shared__ __bf16 Ps[WAVES][QTILE][KCHUNK];  // 8 KB

  const int w    = threadIdx.x >> 5;
  const int lane = threadIdx.x & 31;
  const int blocks_per_bh = SEQ / (QTILE * WAVES);  // 16
  const int bh = blockIdx.x / blocks_per_bh;
  const int q0 = (blockIdx.x % blocks_per_bh) * QTILE * WAVES + w * QTILE;

  const __bf16* Qb = Qd + (size_t)bh * SEQ * DH;
  const __bf16* Kb = Kd + (size_t)bh * SEQ * DH;
  const __bf16* Vb = Vt + (size_t)bh * DH * SEQ;

  const int lr = lane & 15;
  const int hi = lane >> 4;

  v16bf qf[2];
#pragma unroll
  for (int kk = 0; kk < 2; ++kk) {
    const __bf16* qp = Qb + (size_t)(q0 + lr) * DH + kk * 32 + hi * 16;
    qf[kk] = cat16(*(const v8bf*)qp, *(const v8bf*)(qp + 8));
  }

  v8f o[4] = {};                 // 16x64 fp32 accumulator (C-layout)
  float mrow[8], lrow[8];
#pragma unroll
  for (int r = 0; r < 8; ++r) { mrow[r] = -1e30f; lrow[r] = 0.f; }
  const float scale = 0.125f;    // 1/sqrt(64)

  for (int kc = 0; kc < SEQ; kc += KCHUNK) {
    if (kc + KCHUNK < SEQ)
      __builtin_prefetch(Kb + (size_t)(kc + KCHUNK) * DH, 0, 1);

    v8f s[2] = {};
#pragma unroll
    for (int nb = 0; nb < 2; ++nb) {
#pragma unroll
      for (int kk = 0; kk < 2; ++kk) {
        const __bf16* kp =
            Kb + (size_t)(kc + nb * 16 + lr) * DH + kk * 32 + hi * 16;
        v16bf kfrag = cat16(*(const v8bf*)kp, *(const v8bf*)(kp + 8));
        s[nb] = __builtin_amdgcn_wmma_f32_16x16x32_bf16(
            false, qf[kk], false, kfrag, (short)0, s[nb], false, false);
      }
    }

#pragma unroll
    for (int r = 0; r < 8; ++r) {
      float s0 = s[0][r] * scale, s1 = s[1][r] * scale;
      float mx = fmaxf(s0, s1);
#pragma unroll
      for (int off = 1; off < 16; off <<= 1)
        mx = fmaxf(mx, __shfl_xor(mx, off, 32));
      const float mnew = fmaxf(mrow[r], mx);
      const float e0 = __expf(s0 - mnew);
      const float e1 = __expf(s1 - mnew);
      s[0][r] = e0; s[1][r] = e1;
      float sum = e0 + e1;
#pragma unroll
      for (int off = 1; off < 16; off <<= 1)
        sum += __shfl_xor(sum, off, 32);
      const float corr = __expf(mrow[r] - mnew);
      lrow[r] = lrow[r] * corr + sum;
      mrow[r] = mnew;
#pragma unroll
      for (int nb = 0; nb < 4; ++nb) o[nb][r] *= corr;
    }

#pragma unroll
    for (int nb = 0; nb < 2; ++nb)
#pragma unroll
      for (int r = 0; r < 8; ++r)
        Ps[w][hi * 8 + r][nb * 16 + lr] = (__bf16)s[nb][r];

    const __bf16* pp = &Ps[w][lr][hi * 16];
    v16bf pfrag = cat16(*(const v8bf*)pp, *(const v8bf*)(pp + 8));

#pragma unroll
    for (int nb = 0; nb < 4; ++nb) {
      const __bf16* vp = Vb + (size_t)(nb * 16 + lr) * SEQ + kc + hi * 16;
      v16bf vfrag = cat16(*(const v8bf*)vp, *(const v8bf*)(vp + 8));
      o[nb] = __builtin_amdgcn_wmma_f32_16x16x32_bf16(
          false, pfrag, false, vfrag, (short)0, o[nb], false, false);
    }
  }

  const int b = bh / NH, h = bh % NH;
#pragma unroll
  for (int nb = 0; nb < 4; ++nb) {
    const int col = h * DH + nb * 16 + lr;
#pragma unroll
    for (int r = 0; r < 8; ++r) {
      const int row = b * SEQ + q0 + hi * 8 + r;
      Out[(size_t)row * DIM + col] = (__bf16)(o[nb][r] / lrow[r]);
    }
  }
}

// ---------------- launch ----------------

extern "C" void kernel_launch(void* const* d_in, const int* in_sizes, int n_in,
                              void* d_out, int out_size, void* d_ws,
                              size_t ws_size, hipStream_t stream) {
  const float* x     = (const float*)d_in[0];
  const float* W_qkv = (const float*)d_in[1];
  const float* b_qkv = (const float*)d_in[2];
  const float* W_out = (const float*)d_in[3];
  const float* b_out = (const float*)d_in[4];
  float* out = (float*)d_out;

  char* ws = (char*)d_ws;
  size_t off = 0;
  __bf16* Xb    = (__bf16*)(ws + off); off += (size_t)MTOT * DIM * 2;  // reused as attn out
  __bf16* Wqkvt = (__bf16*)(ws + off); off += (size_t)DIM * QKVN * 2;
  __bf16* Woutt = (__bf16*)(ws + off); off += (size_t)DIM * DIM * 2;
  __bf16* Qb    = (__bf16*)(ws + off); off += (size_t)BATCH * NH * SEQ * DH * 2;
  __bf16* Kb    = (__bf16*)(ws + off); off += (size_t)BATCH * NH * SEQ * DH * 2;
  __bf16* Vtb   = (__bf16*)(ws + off); off += (size_t)BATCH * NH * SEQ * DH * 2;
  __bf16* Ob    = Xb;  // x-bf16 dead after QKV GEMM; alias to save workspace
  (void)ws_size; (void)in_sizes; (void)n_in; (void)out_size;

  cvt_f32_bf16<<<2048, 256, 0, stream>>>(x, Xb, MTOT * DIM);
  cvt_w_transpose<<<2048, 256, 0, stream>>>(W_qkv, Wqkvt, DIM, QKVN);
  cvt_w_transpose<<<1024, 256, 0, stream>>>(W_out, Woutt, DIM, DIM);

  // QKV projection GEMM: 4096 x 2304 x 768
  gemm_bf16<<<(MTOT / TM) * (QKVN / TN), 256, 0, stream>>>(
      Xb, Wqkvt, b_qkv, QKVN, 1, nullptr, Qb, Kb, Vtb);

  // flash attention: 24 bh * 16 blocks of 128 q rows
  flash_attn<<<BATCH * NH * (SEQ / (QTILE * WAVES)), 256, 0, stream>>>(
      Qb, Kb, Vtb, Ob);

  // output projection GEMM: 4096 x 768 x 768 -> fp32 d_out
  gemm_bf16<<<(MTOT / TM) * (DIM / TN), 256, 0, stream>>>(
      Ob, Woutt, b_out, DIM, 0, out, nullptr, nullptr, nullptr);
}
